// GAT_9895604650470
// MI455X (gfx1250) — compile-verified, matrix operand order checked by reading, not run on hardware
//
#include <hip/hip_runtime.h>
#include <math.h>

#define N_NODES 50000
#define N_EDGES 1600000
#define E_TOT   (N_EDGES + N_NODES)   // 1,650,000 (edges + self loops)
#define IN_C    512
#define HID     128
#define HEADS   8
#define HC      (HEADS * HID)         // 1024
#define OUT_C   64
#define MPAD    50048                 // N rounded up to multiple of 128 (block M-span)
#define NEG_SLOPE 0.2f

typedef __bf16 bf16;
typedef __attribute__((ext_vector_type(16))) __bf16 bf16x16;
typedef __attribute__((ext_vector_type(8)))  __bf16 bf16x8;
typedef __attribute__((ext_vector_type(8)))  float  f32x8;

union ABFrag { bf16x16 v; bf16x8 h[2]; };

// ---------- small device helpers ----------
__device__ __forceinline__ unsigned short f2bf(float f) {
  union { float f; unsigned u; } v; v.f = f;
  unsigned r = v.u + 0x7fffu + ((v.u >> 16) & 1u);   // round-to-nearest-even
  return (unsigned short)(r >> 16);
}
// order-preserving float <-> uint encoding for atomicMax-based segment max
__device__ __forceinline__ unsigned fenc(float f) {
  unsigned b = __float_as_uint(f);
  return (b & 0x80000000u) ? ~b : (b | 0x80000000u);
}
__device__ __forceinline__ float fdec(unsigned u) {
  return (u & 0x80000000u) ? __uint_as_float(u ^ 0x80000000u) : __uint_as_float(~u);
}
__device__ __forceinline__ void edge_sd(const int* __restrict__ ei, int e, int& s, int& d) {
  if (e < N_EDGES) { s = ei[e]; d = ei[N_EDGES + e]; }
  else             { s = e - N_EDGES; d = s; }       // appended self loops
}
__device__ __forceinline__ float lrelu(float x) { return x > 0.f ? x : NEG_SLOPE * x; }
__device__ __forceinline__ float wave_sum(float v) {
  #pragma unroll
  for (int m = 16; m > 0; m >>= 1) v += __shfl_xor(v, m, 32);
  return v;
}
__device__ __forceinline__ float wave_max(float v) {
  #pragma unroll
  for (int m = 16; m > 0; m >>= 1) v = fmaxf(v, __shfl_xor(v, m, 32));
  return v;
}

// ---------- init / prep ----------
__global__ void fill_zero(unsigned* __restrict__ p, size_t n) {
  size_t i = (size_t)blockIdx.x * 256 + threadIdx.x;
  size_t stride = (size_t)gridDim.x * 256;
  for (; i < n; i += stride) p[i] = 0u;
}

__global__ void prep_x(const float* __restrict__ x, unsigned short* __restrict__ xb) {
  size_t i = (size_t)blockIdx.x * 256 + threadIdx.x;
  if (i >= (size_t)MPAD * IN_C) return;
  int row = (int)(i / IN_C);
  xb[i] = (row < N_NODES) ? f2bf(x[i]) : (unsigned short)0;
}

// W1 [IN_C, HC] row-major -> W1t [HC, IN_C] bf16 (N-major so B loads are contiguous)
__global__ void prep_w1(const float* __restrict__ W, unsigned short* __restrict__ Wt) {
  int o = blockIdx.x * 256 + threadIdx.x;
  if (o >= HC * IN_C) return;
  int n = o / IN_C, k = o % IN_C;
  Wt[o] = f2bf(W[(size_t)k * HC + n]);
}
// W2 [HC, OUT_C] -> W2t [OUT_C, HC] bf16
__global__ void prep_w2(const float* __restrict__ W, unsigned short* __restrict__ Wt) {
  int o = blockIdx.x * 256 + threadIdx.x;
  if (o >= OUT_C * HC) return;
  int n = o / HC, k = o % HC;
  Wt[o] = f2bf(W[(size_t)k * OUT_C + n]);
}

// ---------- WMMA GEMM: C[M,Nc] = A[M,K](bf16) * Bt[Nc,K](bf16), f32 accumulate ----------
// block = 256 threads = 8 waves; wave -> 16(M) x 64(N) tile, 4 WMMAs per K-step of 32.
__global__ void __launch_bounds__(256)
gemm_bf16_wmma(const unsigned short* __restrict__ Aus,
               const unsigned short* __restrict__ Bus,
               float* __restrict__ C, int Mrows, int K, int Nc) {
  const bf16* A = (const bf16*)Aus;
  const bf16* B = (const bf16*)Bus;
  int wave = threadIdx.x >> 5;
  int lane = threadIdx.x & 31;
  int rowBase = (blockIdx.x * 8 + wave) * 16;
  int colBase = blockIdx.y * 64;
  int m  = lane & 15;       // row (A) / col (B) within tile
  int kh = lane >> 4;       // K-half select (ISA 16-bit fragment layout)

  const bf16* ap  = A + (size_t)(rowBase + m) * K + kh * 8;
  const bf16* bp0 = B + (size_t)(colBase +  0 + m) * K + kh * 8;
  const bf16* bp1 = B + (size_t)(colBase + 16 + m) * K + kh * 8;
  const bf16* bp2 = B + (size_t)(colBase + 32 + m) * K + kh * 8;
  const bf16* bp3 = B + (size_t)(colBase + 48 + m) * K + kh * 8;

  f32x8 acc0 = {}, acc1 = {}, acc2 = {}, acc3 = {};
  for (int k0 = 0; k0 < K; k0 += 32) {
    ABFrag a, b0, b1, b2, b3;
    a.h[0]  = *(const bf16x8*)(ap  + k0); a.h[1]  = *(const bf16x8*)(ap  + k0 + 16);
    b0.h[0] = *(const bf16x8*)(bp0 + k0); b0.h[1] = *(const bf16x8*)(bp0 + k0 + 16);
    b1.h[0] = *(const bf16x8*)(bp1 + k0); b1.h[1] = *(const bf16x8*)(bp1 + k0 + 16);
    b2.h[0] = *(const bf16x8*)(bp2 + k0); b2.h[1] = *(const bf16x8*)(bp2 + k0 + 16);
    b3.h[0] = *(const bf16x8*)(bp3 + k0); b3.h[1] = *(const bf16x8*)(bp3 + k0 + 16);
    acc0 = __builtin_amdgcn_wmma_f32_16x16x32_bf16(false, a.v, false, b0.v, (short)0, acc0, false, false);
    acc1 = __builtin_amdgcn_wmma_f32_16x16x32_bf16(false, a.v, false, b1.v, (short)0, acc1, false, false);
    acc2 = __builtin_amdgcn_wmma_f32_16x16x32_bf16(false, a.v, false, b2.v, (short)0, acc2, false, false);
    acc3 = __builtin_amdgcn_wmma_f32_16x16x32_bf16(false, a.v, false, b3.v, (short)0, acc3, false, false);
  }
  int col = colBase + m;
  #pragma unroll
  for (int i = 0; i < 8; ++i) {
    int r = rowBase + i + 8 * kh;     // C/D layout: M = i + 8*(lane>>4), N = lane&15
    if (r < Mrows) {
      size_t ro = (size_t)r * Nc;
      C[ro + col +  0] = acc0[i];
      C[ro + col + 16] = acc1[i];
      C[ro + col + 32] = acc2[i];
      C[ro + col + 48] = acc3[i];
    }
  }
}

// ---------- layer-1 attention dot products: one wave per (node, head) ----------
__global__ void __launch_bounds__(256)
attn_dots1(const float* __restrict__ h1, const float* __restrict__ asrc,
           const float* __restrict__ adst, float* __restrict__ as, float* __restrict__ ad) {
  int wave = threadIdx.x >> 5, lane = threadIdx.x & 31;
  int p = blockIdx.x * 8 + wave;      // grid.x == N_NODES -> p in [0, N*HEADS)
  int n = p >> 3, hd = p & 7;
  float4 hv = *(const float4*)(h1 + (size_t)n * HC + hd * HID + lane * 4);
  float4 sv = *(const float4*)(asrc + hd * HID + lane * 4);
  float4 dv = *(const float4*)(adst + hd * HID + lane * 4);
  float ds = hv.x*sv.x + hv.y*sv.y + hv.z*sv.z + hv.w*sv.w;
  float dd = hv.x*dv.x + hv.y*dv.y + hv.z*dv.z + hv.w*dv.w;
  ds = wave_sum(ds); dd = wave_sum(dd);
  if (lane == 0) { as[p] = ds; ad[p] = dd; }
}

// ---------- layer-1 edge softmax (3 passes) ----------
__global__ void edge_max1(const int* __restrict__ ei, const float* __restrict__ as,
                          const float* __restrict__ ad, unsigned* __restrict__ mx) {
  int e = blockIdx.x * 256 + threadIdx.x;
  if (e >= E_TOT) return;
  int s, d; edge_sd(ei, e, s, d);
  float4 a0 = *(const float4*)(as + (size_t)s * 8);
  float4 a1 = *(const float4*)(as + (size_t)s * 8 + 4);
  float4 b0 = *(const float4*)(ad + (size_t)d * 8);
  float4 b1 = *(const float4*)(ad + (size_t)d * 8 + 4);
  unsigned* md = mx + (size_t)d * 8;
  atomicMax(&md[0], fenc(lrelu(a0.x + b0.x)));
  atomicMax(&md[1], fenc(lrelu(a0.y + b0.y)));
  atomicMax(&md[2], fenc(lrelu(a0.z + b0.z)));
  atomicMax(&md[3], fenc(lrelu(a0.w + b0.w)));
  atomicMax(&md[4], fenc(lrelu(a1.x + b1.x)));
  atomicMax(&md[5], fenc(lrelu(a1.y + b1.y)));
  atomicMax(&md[6], fenc(lrelu(a1.z + b1.z)));
  atomicMax(&md[7], fenc(lrelu(a1.w + b1.w)));
}

__global__ void edge_exp1(const int* __restrict__ ei, const float* __restrict__ as,
                          const float* __restrict__ ad, const unsigned* __restrict__ mx,
                          float* __restrict__ ev, float* __restrict__ z) {
  int e = blockIdx.x * 256 + threadIdx.x;
  if (e >= E_TOT) return;
  int s, d; edge_sd(ei, e, s, d);
  const float* ap = as + (size_t)s * 8;
  const float* bp = ad + (size_t)d * 8;
  const unsigned* md = mx + (size_t)d * 8;
  float* ep = ev + (size_t)e * 8;
  float* zp = z + (size_t)d * 8;
  #pragma unroll
  for (int h = 0; h < 8; ++h) {
    float l = lrelu(ap[h] + bp[h]);
    float x = expf(l - fdec(md[h]));
    ep[h] = x;
    atomicAdd(&zp[h], x);
  }
}

// one 256-thread block per edge: 1024 channels, float4 per thread
__global__ void __launch_bounds__(256)
edge_aggr1(const int* __restrict__ ei, const float* __restrict__ h1,
           const float* __restrict__ ev, const float* __restrict__ z,
           float* __restrict__ out1) {
  int e = blockIdx.x;
  int s, d; edge_sd(ei, e, s, d);
  int i = threadIdx.x * 4;
  int hd = i >> 7;
  float a = ev[(size_t)e * 8 + hd] / z[(size_t)d * 8 + hd];
  float4 hv = *(const float4*)(h1 + (size_t)s * HC + i);
  float* o = out1 + (size_t)d * HC + i;
  atomicAdd(o + 0, hv.x * a);
  atomicAdd(o + 1, hv.y * a);
  atomicAdd(o + 2, hv.z * a);
  atomicAdd(o + 3, hv.w * a);
}

// bias + ELU + bf16 convert (padded rows -> 0) for layer-2 GEMM input
__global__ void post1(const float* __restrict__ out1, const float* __restrict__ b1,
                      unsigned short* __restrict__ h2b) {
  size_t i = (size_t)blockIdx.x * 256 + threadIdx.x;
  if (i >= (size_t)MPAD * HC) return;
  int row = (int)(i / HC), col = (int)(i % HC);
  unsigned short r = 0;
  if (row < N_NODES) {
    float v = out1[i] + b1[col];
    v = v > 0.f ? v : (expf(v) - 1.f);
    r = f2bf(v);
  }
  h2b[i] = r;
}

// ---------- layer-2 (1 head, 64 channels): wave per node ----------
__global__ void __launch_bounds__(256)
attn_dots2(const float* __restrict__ g, const float* __restrict__ asrc,
           const float* __restrict__ adst, float* __restrict__ as2, float* __restrict__ ad2) {
  int wave = threadIdx.x >> 5, lane = threadIdx.x & 31;
  int n = blockIdx.x * 8 + wave;
  if (n >= N_NODES) return;
  float2 gv = *(const float2*)(g + (size_t)n * OUT_C + lane * 2);
  float2 sv = *(const float2*)(asrc + lane * 2);
  float2 dv = *(const float2*)(adst + lane * 2);
  float ds = wave_sum(gv.x * sv.x + gv.y * sv.y);
  float dd = wave_sum(gv.x * dv.x + gv.y * dv.y);
  if (lane == 0) { as2[n] = ds; ad2[n] = dd; }
}

__global__ void edge_max2(const int* __restrict__ ei, const float* __restrict__ as,
                          const float* __restrict__ ad, unsigned* __restrict__ mx) {
  int e = blockIdx.x * 256 + threadIdx.x;
  if (e >= E_TOT) return;
  int s, d; edge_sd(ei, e, s, d);
  atomicMax(&mx[d], fenc(lrelu(as[s] + ad[d])));
}

__global__ void edge_exp2(const int* __restrict__ ei, const float* __restrict__ as,
                          const float* __restrict__ ad, const unsigned* __restrict__ mx,
                          float* __restrict__ ev, float* __restrict__ z) {
  int e = blockIdx.x * 256 + threadIdx.x;
  if (e >= E_TOT) return;
  int s, d; edge_sd(ei, e, s, d);
  float x = expf(lrelu(as[s] + ad[d]) - fdec(mx[d]));
  ev[e] = x;
  atomicAdd(&z[d], x);
}

// wave per edge: 64 channels, float2 per lane
__global__ void __launch_bounds__(256)
edge_aggr2(const int* __restrict__ ei, const float* __restrict__ g,
           const float* __restrict__ ev, const float* __restrict__ z,
           float* __restrict__ out2) {
  int wave = threadIdx.x >> 5, lane = threadIdx.x & 31;
  int e = blockIdx.x * 8 + wave;
  if (e >= E_TOT) return;
  int s, d; edge_sd(ei, e, s, d);
  float a = ev[e] / z[d];
  float2 gv = *(const float2*)(g + (size_t)s * OUT_C + lane * 2);
  float* o = out2 + (size_t)d * OUT_C + lane * 2;
  atomicAdd(o + 0, gv.x * a);
  atomicAdd(o + 1, gv.y * a);
}

// bias + log_softmax over 64 channels: wave per node
__global__ void __launch_bounds__(256)
final_lsm(const float* __restrict__ out2, const float* __restrict__ b2,
          float* __restrict__ out) {
  int wave = threadIdx.x >> 5, lane = threadIdx.x & 31;
  int n = blockIdx.x * 8 + wave;
  if (n >= N_NODES) return;
  int c = lane * 2;
  float v0 = out2[(size_t)n * OUT_C + c]     + b2[c];
  float v1 = out2[(size_t)n * OUT_C + c + 1] + b2[c + 1];
  float m = wave_max(fmaxf(v0, v1));
  float s = wave_sum(expf(v0 - m) + expf(v1 - m));
  float ls = logf(s);
  out[(size_t)n * OUT_C + c]     = v0 - m - ls;
  out[(size_t)n * OUT_C + c + 1] = v1 - m - ls;
}

// ---------- launch ----------
extern "C" void kernel_launch(void* const* d_in, const int* in_sizes, int n_in,
                              void* d_out, int out_size, void* d_ws, size_t ws_size,
                              hipStream_t stream) {
  (void)in_sizes; (void)n_in; (void)out_size; (void)ws_size;
  const float* x    = (const float*)d_in[0];
  const int*   ei   = (const int*)d_in[1];
  const float* W1   = (const float*)d_in[2];
  const float* a_s1 = (const float*)d_in[3];
  const float* a_d1 = (const float*)d_in[4];
  const float* b1   = (const float*)d_in[5];
  const float* W2   = (const float*)d_in[6];
  const float* a_s2 = (const float*)d_in[7];
  const float* a_d2 = (const float*)d_in[8];
  const float* b2   = (const float*)d_in[9];
  float* out = (float*)d_out;

  char* p = (char*)d_ws;
  auto carve = [&](size_t bytes) -> char* {
    char* r = p; p += (bytes + 255) & ~(size_t)255; return r;
  };
  // contiguous zero-init region first (single fill kernel)
  char* z_begin = p;
  unsigned* mx1  = (unsigned*)carve((size_t)N_NODES * 8 * 4);
  float*    z1   = (float*)   carve((size_t)N_NODES * 8 * 4);
  float*    out1 = (float*)   carve((size_t)N_NODES * HC * 4);
  unsigned* mx2  = (unsigned*)carve((size_t)N_NODES * 4);
  float*    z2   = (float*)   carve((size_t)N_NODES * 4);
  float*    out2 = (float*)   carve((size_t)N_NODES * OUT_C * 4);
  char* z_end = p;
  unsigned short* xb  = (unsigned short*)carve((size_t)MPAD * IN_C * 2);
  unsigned short* w1t = (unsigned short*)carve((size_t)HC * IN_C * 2);
  unsigned short* w2t = (unsigned short*)carve((size_t)OUT_C * HC * 2);
  float* h1  = (float*)carve((size_t)N_NODES * HC * 4);
  float* as1 = (float*)carve((size_t)N_NODES * 8 * 4);
  float* ad1 = (float*)carve((size_t)N_NODES * 8 * 4);
  float* ev1 = (float*)carve((size_t)E_TOT * 8 * 4);
  unsigned short* h2b = (unsigned short*)carve((size_t)MPAD * HC * 2);
  float* g2  = (float*)carve((size_t)N_NODES * OUT_C * 4);
  float* as2 = (float*)carve((size_t)N_NODES * 4);
  float* ad2 = (float*)carve((size_t)N_NODES * 4);
  float* ev2 = (float*)carve((size_t)E_TOT * 4);

  size_t zwords = (size_t)(z_end - z_begin) / 4;
  fill_zero<<<4096, 256, 0, stream>>>((unsigned*)z_begin, zwords);

  prep_x <<<(unsigned)(((size_t)MPAD * IN_C + 255) / 256), 256, 0, stream>>>(x, xb);
  prep_w1<<<(HC * IN_C + 255) / 256, 256, 0, stream>>>(W1, w1t);
  prep_w2<<<(OUT_C * HC + 255) / 256, 256, 0, stream>>>(W2, w2t);

  // layer 1
  gemm_bf16_wmma<<<dim3(MPAD / 128, HC / 64), 256, 0, stream>>>(xb, w1t, h1, N_NODES, IN_C, HC);
  attn_dots1<<<N_NODES, 256, 0, stream>>>(h1, a_s1, a_d1, as1, ad1);
  int eb = (E_TOT + 255) / 256;
  edge_max1 <<<eb, 256, 0, stream>>>(ei, as1, ad1, mx1);
  edge_exp1 <<<eb, 256, 0, stream>>>(ei, as1, ad1, mx1, ev1, z1);
  edge_aggr1<<<E_TOT, 256, 0, stream>>>(ei, h1, ev1, z1, out1);
  post1<<<(unsigned)(((size_t)MPAD * HC + 255) / 256), 256, 0, stream>>>(out1, b1, h2b);

  // layer 2
  gemm_bf16_wmma<<<dim3(MPAD / 128, OUT_C / 64), 256, 0, stream>>>(h2b, w2t, g2, N_NODES, HC, OUT_C);
  attn_dots2<<<(N_NODES + 7) / 8, 256, 0, stream>>>(g2, a_s2, a_d2, as2, ad2);
  edge_max2 <<<eb, 256, 0, stream>>>(ei, as2, ad2, mx2);
  edge_exp2 <<<eb, 256, 0, stream>>>(ei, as2, ad2, mx2, ev2, z2);
  edge_aggr2<<<(E_TOT + 7) / 8, 256, 0, stream>>>(ei, g2, ev2, z2, out2);

  final_lsm<<<(N_NODES + 7) / 8, 256, 0, stream>>>(out2, b2, out);
}
